// QCCNN_64948495450125
// MI455X (gfx1250) — matrix operand order repaired and checked by reading, not compiled, not run on hardware
//
#include <hip/hip_runtime.h>
#include <hip/hip_bf16.h>

typedef __attribute__((ext_vector_type(16))) _Float16 v16h;
typedef __attribute__((ext_vector_type(8)))  float    v8f;

#define N_KER 4
#define WAVES_PER_BLOCK 9
#define THREADS (WAVES_PER_BLOCK * 32)

__device__ __forceinline__ float leaky_f(float v) { return v >= 0.f ? v : 0.1f * v; }

// ---------------------------------------------------------------------------
// Kernel 0: fold the 4-qubit circuit (Rot gates + CNOT chain) into a complex
// 16x16 matrix per quantum kernel.  Stored as B[j][i] = M[i][j] (re then im),
// which is exactly the WMMA B-fragment order used by the main kernel.
// ---------------------------------------------------------------------------
__global__ void qccnn_prep(const float* __restrict__ qw, float* __restrict__ Bws) {
  int t = threadIdx.x;
  if (t >= 64) return;
  int k = t >> 4, i = t & 15;

  float Ur[4][2][2], Ui[4][2][2];
  for (int q = 0; q < 4; ++q) {
    float phi = qw[(k * 4 + q) * 3 + 0];
    float th  = qw[(k * 4 + q) * 3 + 1];
    float om  = qw[(k * 4 + q) * 3 + 2];
    float c = cosf(0.5f * th), s = sinf(0.5f * th);
    float ap = -0.5f * (phi + om);            // ep = exp(i*ap)
    float am =  0.5f * (phi - om);            // em = exp(i*am)
    float epr = cosf(ap), epi = sinf(ap);
    float emr = cosf(am), emi = sinf(am);
    Ur[q][0][0] =  epr * c; Ui[q][0][0] =  epi * c;   // m00 = ep*c
    Ur[q][0][1] = -emr * s; Ui[q][0][1] = -emi * s;   // m01 = -em*s
    Ur[q][1][0] =  emr * s; Ui[q][1][0] = -emi * s;   // m10 = conj(em)*s
    Ur[q][1][1] =  epr * c; Ui[q][1][1] = -epi * c;   // m11 = conj(ep)*c
  }

  // CNOT-chain permutation of output index i (b0 = qubit0 = MSB)
  int b0 = (i >> 3) & 1, b1 = (i >> 2) & 1, b2 = (i >> 1) & 1, b3 = i & 1;
  if (b0) b1 ^= 1;
  if (b1) b2 ^= 1;
  if (b2) b3 ^= 1;
  if (b3) b0 ^= 1;
  int ip = b0 * 8 + b1 * 4 + b2 * 2 + b3;

  int ib[4] = { (i >> 3) & 1, (i >> 2) & 1, (i >> 1) & 1, i & 1 };
  for (int j = 0; j < 16; ++j) {
    int jb[4] = { (j >> 3) & 1, (j >> 2) & 1, (j >> 1) & 1, j & 1 };
    float tr = 1.f, ti = 0.f;
    for (int q = 0; q < 4; ++q) {
      float ur = Ur[q][ib[q]][jb[q]], ui = Ui[q][ib[q]][jb[q]];
      float nr = tr * ur - ti * ui;
      float ni = tr * ui + ti * ur;
      tr = nr; ti = ni;
    }
    Bws[(k * 2 + 0) * 256 + j * 16 + ip] = tr;   // B_re[j][ip] = M_re[ip][j]
    Bws[(k * 2 + 1) * 256 + j * 16 + ip] = ti;   // B_im[j][ip] = M_im[ip][j]
  }
}

// ---------------------------------------------------------------------------
// Main fused kernel: 16 batch elements / block (= 144 patches = 9 WMMA tiles,
// one per wave).  Quantum mat-vecs and FC1 are done with
// v_wmma_f32_16x16x32_f16; FC2 (32->3) is a tiny per-lane dot product.
// ---------------------------------------------------------------------------
__global__ __launch_bounds__(THREADS) void qccnn_main(
    const float* __restrict__ x,
    const float* __restrict__ fc1_w, const float* __restrict__ fc1_b,
    const float* __restrict__ fc2_w, const float* __restrict__ fc2_b,
    const float* __restrict__ Bws, float* __restrict__ out)
{
  __shared__ float ampsS[WAVES_PER_BLOCK][16][17];   // raw patches
  __shared__ float vreS [WAVES_PER_BLOCK][16][17];   // WMMA state output (re)
  __shared__ float vimS [WAVES_PER_BLOCK][16][17];   // WMMA state output (im)
  __shared__ float hfeatS[16][128];                  // leaky(features), padded
  __shared__ float h2S[16][33];                      // FC1 activations

  const int tid  = threadIdx.x;
  const int wv   = tid >> 5;
  const int lane = tid & 31;
  const long b0  = (long)blockIdx.x * 16;

  // zero hfeat: provides the K=108..127 zero padding for the FC1 WMMA
  for (int f = tid; f < 16 * 128; f += THREADS) hfeatS[f >> 7][f & 127] = 0.f;

  // load this wave's 16 patches (stride-2 4x4 windows) into LDS
  const int pbase = wv * 16;
#pragma unroll
  for (int t = 0; t < 8; ++t) {
    int f  = t * 32 + lane;                 // 0..255 flat (patch, elem)
    int pl = f >> 4, e = f & 15;
    int pg = pbase + pl;                    // local patch id 0..143
    int bl = pg / 9, cell = pg % 9;
    int hh = cell / 3, ww = cell % 3;
    int r = e >> 2, c = e & 3;
    ampsS[wv][pl][e] = x[(b0 + bl) * 64 + (hh * 2 + r) * 8 + (ww * 2 + c)];
  }
  __syncthreads();

  const int m  = lane & 15;   // A-frag row / D-frag column
  const int hi = lane >> 4;

  // reciprocal norm of patch m (duplicated across lane halves — trivial work)
  float ss = 0.f;
#pragma unroll
  for (int e = 0; e < 16; ++e) { float v = ampsS[wv][m][e]; ss += v * v; }
  const float rn = 1.0f / (sqrtf(ss) + 1e-12f);

  // A fragment: A[m][K] with K = hi*8+h in halves 0..7, zero for K>=16
  v16h afrag;
#pragma unroll
  for (int h = 0; h < 8; ++h) {
    afrag[h]     = (_Float16)(ampsS[wv][m][hi * 8 + h] * rn);
    afrag[h + 8] = (_Float16)0.f;
  }

  const int pg0   = pbase + m;       // feature patch for lanes < 16
  const int bl0   = pg0 / 9;
  const int cell0 = pg0 % 9;

#pragma unroll
  for (int k = 0; k < N_KER; ++k) {
    // B fragments: lane = state column i (m), halves = K row j = hi*8+n, zero for K>=16
    v16h bre, bim;
#pragma unroll
    for (int n = 0; n < 16; ++n) {
      float vr = (n < 8) ? Bws[(k * 2 + 0) * 256 + (hi * 8 + n) * 16 + m] : 0.f;
      float vi = (n < 8) ? Bws[(k * 2 + 1) * 256 + (hi * 8 + n) * 16 + m] : 0.f;
      bre[n] = (_Float16)vr;
      bim[n] = (_Float16)vi;
    }
    v8f cz = {};
    v8f dre = __builtin_amdgcn_wmma_f32_16x16x32_f16(false, afrag, false, bre,
                                                     (short)0, cz, false, false);
    v8f dim = __builtin_amdgcn_wmma_f32_16x16x32_f16(false, afrag, false, bim,
                                                     (short)0, cz, false, false);
    // D: row = r + 8*hi (patch), col = m (state index) -> LDS
#pragma unroll
    for (int r = 0; r < 8; ++r) {
      vreS[wv][r + 8 * hi][m] = dre[r];
      vimS[wv][r + 8 * hi][m] = dim[r];
    }
    __builtin_amdgcn_wave_barrier();      // same-wave LDS is in-order
    if (lane < 16) {
      float X = 0.f, Y = 0.f, Z = 0.f;
#pragma unroll
      for (int j = 0; j < 8; ++j) {
        float ar = vreS[wv][m][j],     ai = vimS[wv][m][j];
        float br = vreS[wv][m][j + 8], bi = vimS[wv][m][j + 8];
        X += ar * br + ai * bi;           // 2*Re(conj(a)b)
        Y += ar * bi - ai * br;           // 2*Im(conj(a)b)
        Z += ar * ar + ai * ai - br * br - bi * bi;
      }
      X *= 2.f; Y *= 2.f;
      hfeatS[bl0][(3 * k + 0) * 9 + cell0] = leaky_f(X);
      hfeatS[bl0][(3 * k + 1) * 9 + cell0] = leaky_f(Y);
      hfeatS[bl0][(3 * k + 2) * 9 + cell0] = leaky_f(Z);
    }
    __builtin_amdgcn_wave_barrier();
  }

  __syncthreads();

  if (wv == 0) {
    // FC1: h(16x128, zero padded from 108) x W1^T -> 16x32, via WMMA
    v16h afc[4];
#pragma unroll
    for (int c = 0; c < 4; ++c) {
#pragma unroll
      for (int h = 0; h < 8; ++h) {
        afc[c][h]     = (_Float16)hfeatS[m][c * 32 + hi * 8 + h];
        afc[c][h + 8] = (_Float16)hfeatS[m][c * 32 + 16 + hi * 8 + h];
      }
    }
#pragma unroll
    for (int nt = 0; nt < 2; ++nt) {
      v8f acc = {};
#pragma unroll
      for (int c = 0; c < 4; ++c) {
        v16h bfr;
#pragma unroll
        for (int n = 0; n < 16; ++n) {
          int j = c * 32 + (n < 8 ? hi * 8 + n : 16 + hi * 8 + (n - 8));
          bfr[n] = (_Float16)((j < 108) ? fc1_w[(nt * 16 + m) * 108 + j] : 0.f);
        }
        acc = __builtin_amdgcn_wmma_f32_16x16x32_f16(false, afc[c], false, bfr,
                                                     (short)0, acc, false, false);
      }
      int i = nt * 16 + m;
      float bias = fc1_b[i];
#pragma unroll
      for (int r = 0; r < 8; ++r)
        h2S[r + 8 * hi][i] = leaky_f(acc[r] + bias);
    }
    __builtin_amdgcn_wave_barrier();
    if (lane < 16) {
#pragma unroll
      for (int o = 0; o < 3; ++o) {
        float s = fc2_b[o];
#pragma unroll
        for (int i2 = 0; i2 < 32; ++i2) s += fc2_w[o * 32 + i2] * h2S[lane][i2];
        out[(b0 + lane) * 3 + o] = s;
      }
    }
  }
}

extern "C" void kernel_launch(void* const* d_in, const int* in_sizes, int n_in,
                              void* d_out, int out_size, void* d_ws, size_t ws_size,
                              hipStream_t stream) {
  const float* x     = (const float*)d_in[0];
  const float* qw    = (const float*)d_in[1];
  const float* fc1_w = (const float*)d_in[2];
  const float* fc1_b = (const float*)d_in[3];
  const float* fc2_w = (const float*)d_in[4];
  const float* fc2_b = (const float*)d_in[5];
  float* out = (float*)d_out;
  float* Bws = (float*)d_ws;     // 4 kernels * {re,im} * 16*16 floats = 8 KB

  int B = in_sizes[0] / 64;      // x is (B,1,8,8)

  qccnn_prep<<<1, 64, 0, stream>>>(qw, Bws);
  qccnn_main<<<B / 16, THREADS, 0, stream>>>(x, fc1_w, fc1_b, fc2_w, fc2_b, Bws, out);
}
